// SparseCompactor_37864431681787
// MI455X (gfx1250) — matrix operand order repaired and checked by reading, not compiled, not run on hardware
//
#include <hip/hip_runtime.h>
#include <math.h>

// ---------------------------------------------------------------------------
// SparseCompactor for MI455X (gfx1250, wave32, WMMA)
//   x:[8,4096,2048] f32, W1:[2048,512], b1:[512], W2:[512,1], b2:[1]
//   importance = (GELU(x@W1+b1) @ W2 + b2); top-k(50%) per batch; gather rows.
// ---------------------------------------------------------------------------

#define BB   8
#define SS   4096
#define HH   2048
#define HQ   512
#define KEEP 2048

typedef __attribute__((ext_vector_type(16))) __bf16 v16bf;
typedef __attribute__((ext_vector_type(8)))  __bf16 v8bf;
typedef __attribute__((ext_vector_type(4)))  __bf16 v4bf;
typedef __attribute__((ext_vector_type(8)))  float  v8f;

// ---------------------------------------------------------------------------
// Kernel 1: W1 [H][Hq] f32  ->  W1t [Hq][H] bf16  (transposed, L2-resident)
// ---------------------------------------------------------------------------
__global__ void k_w1(const float* __restrict__ W1, unsigned short* __restrict__ w1t_u) {
    __bf16* w1t = (__bf16*)w1t_u;
    int g = blockIdx.x * blockDim.x + threadIdx.x;   // 0 .. Hq*H-1
    int c = g >> 11;          // output row    (0..511)
    int k = g & (HH - 1);     // output column (0..2047)
    float f = W1[(size_t)k * HQ + c];
    w1t[g] = (__bf16)f;       // coalesced bf16 writes
}

// ---------------------------------------------------------------------------
// Kernel 2: importance scores via WMMA bf16 (f32 accumulate), fused GELU + W2
//   grid: B * (S/32) blocks of 256 threads (8 waves).
//   LDS: x tile [32 tokens][2048] bf16 = 128 KB (dynamic) + 32-float combine.
//   wave w: M-tile = w&1 (16 tokens), N-group = w>>1 (8 N-tiles of 16 cols).
// ---------------------------------------------------------------------------
__global__ void k_scores(const float* __restrict__ x,
                         const unsigned short* __restrict__ w1t_u,
                         const float* __restrict__ b1,
                         const float* __restrict__ W2,
                         const float* __restrict__ b2,
                         float* __restrict__ scores) {
    extern __shared__ __align__(16) char smem_raw[];
    __bf16* xtile = (__bf16*)smem_raw;                         // [32][2048]
    float*  imp   = (float*)(smem_raw + 32 * HH * sizeof(__bf16));
    const __bf16* w1t = (const __bf16*)w1t_u;

    const int tid  = threadIdx.x;
    const int b    = blockIdx.x >> 7;            // batch
    const int m0   = (blockIdx.x & 127) * 32;    // first token of tile

    if (tid < 32) imp[tid] = 0.0f;

    // ---- stage x[32, 2048] f32 -> bf16 in LDS (coalesced float4 reads) ----
    const float* xg = x + (((size_t)b * SS + m0) << 11);
    #pragma unroll 4
    for (int it = 0; it < 64; ++it) {
        int i4   = it * 256 + tid;               // float4 index in tile
        int elem = i4 << 2;
        float4 v = ((const float4*)xg)[i4];
        __builtin_prefetch(&((const float4*)xg)[i4 + 1024], 0, 1);
        v4bf p;
        p[0] = (__bf16)v.x; p[1] = (__bf16)v.y;
        p[2] = (__bf16)v.z; p[3] = (__bf16)v.w;
        *(v4bf*)(xtile + elem) = p;
    }
    __syncthreads();

    // ---- WMMA: each wave -> 16 tokens x 128 columns ----
    const int wave  = tid >> 5;
    const int lane  = tid & 31;
    const int mtile = wave & 1;                  // 0/1 -> tokens [mtile*16 ..)
    const int ngrp  = wave >> 1;                 // 0..3 -> cols [ngrp*128 ..)
    const int hi    = (lane >= 16);
    const int arow  = mtile * 16 + (lane & 15);  // A-fragment source row
    const __bf16* xrow = xtile + (size_t)arow * HH;
    const int aoff  = hi ? 16 : 0;               // K sub-block select (A layout)
    const int boff  = hi ? 16 : 0;               // K sub-block select (B layout)

    float imp_p[8];
    #pragma unroll
    for (int r = 0; r < 8; ++r) imp_p[r] = 0.0f;

    for (int nt = 0; nt < 8; ++nt) {
        const int nbase = ngrp * 128 + nt * 16;
        const int ncol  = nbase + (lane & 15);   // this lane's N in D-fragment
        const __bf16* bcol = w1t + (size_t)ncol * HH;

        v8f acc = {};
        #pragma unroll 4
        for (int kb = 0; kb < HH; kb += 32) {
            // A: 16x32 bf16 — lane<16: K kb..kb+15; lane>=16: +16
            v8bf lo = *(const v8bf*)(xrow + kb + aoff);
            v8bf hh = *(const v8bf*)(xrow + kb + 8 + aoff);
            v16bf a;
            #pragma unroll
            for (int t = 0; t < 8; ++t) { a[t] = lo[t]; a[t + 8] = hh[t]; }
            // B: 32x16 bf16 — lane<16: K kb..kb+15 at col N=lane; lane>=16: +16
            v16bf bm = *(const v16bf*)(bcol + kb + boff);
            acc = __builtin_amdgcn_wmma_f32_16x16x32_bf16(
                false, a, false, bm, (short)0, acc, false, false);
        }

        // D layout: VGPR r, lane L -> M = r + 8*(L>=16), N = L&15
        const float w2c = W2[ncol];
        const float bias = b1[ncol];
        #pragma unroll
        for (int r = 0; r < 8; ++r) {
            float h = acc[r] + bias;
            float g = 0.5f * h * (1.0f + erff(h * 0.70710678118654752f));
            imp_p[r] += g * w2c;
        }
    }

    // reduce over the 16 lanes sharing one token (xor masks 1..8 stay in-half)
    #pragma unroll
    for (int r = 0; r < 8; ++r) {
        float v = imp_p[r];
        v += __shfl_xor(v, 1, 32);
        v += __shfl_xor(v, 2, 32);
        v += __shfl_xor(v, 4, 32);
        v += __shfl_xor(v, 8, 32);
        imp_p[r] = v;
    }
    __syncthreads();   // imp[] zero-init visible
    if (lane == 0 || lane == 16) {
        #pragma unroll
        for (int r = 0; r < 8; ++r) {
            int tl = mtile * 16 + r + (hi ? 8 : 0);
            atomicAdd(&imp[tl], imp_p[r]);   // combine the 4 N-groups
        }
    }
    __syncthreads();
    if (tid < 32)
        scores[(size_t)b * SS + m0 + tid] = imp[tid] + b2[0];
}

// ---------------------------------------------------------------------------
// Kernel 3: per-batch top-K via bitonic sort in LDS (1024 thr, 4096 elems),
// tie-break = lower index first (matches lax.top_k), then re-sort kept
// indices ascending.
// ---------------------------------------------------------------------------
__global__ void k_topk(const float* __restrict__ scores, int* __restrict__ idx_out) {
    __shared__ float skey[SS];
    __shared__ int   sval[SS];
    const int tid = threadIdx.x;
    const int b   = blockIdx.x;
    const float* sb = scores + (size_t)b * SS;

    #pragma unroll
    for (int s = 0; s < 4; ++s) {
        int i = tid + s * 1024;
        skey[i] = sb[i];
        sval[i] = i;
    }
    __syncthreads();

    // sort by (score desc, index asc)
    for (int k = 2; k <= SS; k <<= 1) {
        for (int j = k >> 1; j > 0; j >>= 1) {
            #pragma unroll
            for (int s = 0; s < 4; ++s) {
                int i = tid + s * 1024;
                int p = i ^ j;
                if (p > i) {
                    float ki = skey[i], kp = skey[p];
                    int   vi = sval[i], vp = sval[p];
                    bool up = ((i & k) == 0);
                    bool ib = (ki > kp) || (ki == kp && vi < vp); // i before p
                    if (ib != up) {
                        skey[i] = kp; skey[p] = ki;
                        sval[i] = vp; sval[p] = vi;
                    }
                }
            }
            __syncthreads();
        }
    }

    // kept indices = sval[0..KEEP-1]; sort them ascending
    for (int k = 2; k <= KEEP; k <<= 1) {
        for (int j = k >> 1; j > 0; j >>= 1) {
            #pragma unroll
            for (int s = 0; s < 2; ++s) {
                int i = tid + s * 1024;
                int p = i ^ j;
                if (p > i) {          // i,p < 2048 automatically (k<=2048)
                    int vi = sval[i], vp = sval[p];
                    bool up = ((i & k) == 0);
                    bool ib = (vi < vp);
                    if (ib != up) { sval[i] = vp; sval[p] = vi; }
                }
            }
            __syncthreads();
        }
    }

    #pragma unroll
    for (int s = 0; s < 2; ++s)
        idx_out[(size_t)b * KEEP + tid + s * 1024] = sval[tid + s * 1024];
}

// ---------------------------------------------------------------------------
// Kernel 4: gather kept rows (bandwidth kernel: pure b128 stream)
//   one block per kept token; 2048 floats = 512 float4 / 256 threads.
// ---------------------------------------------------------------------------
__global__ void k_gather(const float* __restrict__ x,
                         const int* __restrict__ idx,
                         float* __restrict__ out) {
    const int b  = blockIdx.x >> 11;
    const int kk = blockIdx.x & (KEEP - 1);
    const int src = idx[((size_t)b << 11) + kk];
    const float4* xs = (const float4*)(x + (((size_t)b * SS + src) << 11));
    float4* od = (float4*)(out + (((size_t)b * KEEP + kk) << 11));
    const int t = threadIdx.x;
    od[t]       = xs[t];
    od[t + 256] = xs[t + 256];
}

// ---------------------------------------------------------------------------
extern "C" void kernel_launch(void* const* d_in, const int* in_sizes, int n_in,
                              void* d_out, int out_size, void* d_ws, size_t ws_size,
                              hipStream_t stream) {
    (void)in_sizes; (void)n_in; (void)out_size; (void)ws_size;
    const float* x  = (const float*)d_in[0];
    const float* W1 = (const float*)d_in[1];
    const float* b1 = (const float*)d_in[2];
    const float* W2 = (const float*)d_in[3];
    const float* b2 = (const float*)d_in[4];
    float* out = (float*)d_out;

    char* ws = (char*)d_ws;
    unsigned short* w1t = (unsigned short*)ws;                       // 2 MB bf16
    float* scores = (float*)(ws + (size_t)HQ * HH * 2);              // 128 KB
    int*   idx    = (int*)(ws + (size_t)HQ * HH * 2 + BB * SS * 4);  // 64 KB

    // 1) W1 -> bf16 transpose (L2-resident operand for the scorer GEMM)
    k_w1<<<(HQ * HH) / 256, 256, 0, stream>>>(W1, w1t);

    // 2) scores: WMMA bf16 scorer, 32 tokens/block staged in 128 KB LDS
    size_t shbytes = (size_t)32 * HH * 2 + 32 * sizeof(float);
    k_scores<<<BB * (SS / 32), 256, shbytes, stream>>>(x, w1t, b1, W2, b2, scores);

    // 3) per-batch top-k + ascending index sort
    k_topk<<<BB, 1024, 0, stream>>>(scores, idx);

    // 4) gather kept tokens (memory-bound streaming copy)
    k_gather<<<BB * KEEP, 256, 0, stream>>>(x, idx, out);
}
